// ContrastiveLoss_20572893348484
// MI455X (gfx1250) — matrix operand order repaired and checked by reading, not compiled, not run on hardware
//
#include <hip/hip_runtime.h>

// Contrastive loss, fused symmetric WMMA formulation for MI455X (gfx1250, wave32).
// sim = normalize(X) @ normalize(X)^T is never materialized. Each wave computes
// a 32x32 macro-tile (2x2 register-blocked 16x16 WMMA tiles) of the upper
// triangle only; symmetric contributions are mirrored into the column rows.
// Main loop: 8 global_load_b128 + 4 v_wmma per k-step, rolled (no spills).
// Epilogue: max3 threshold identity + ballot/popcount for validity counts.

#define UPTH 0.55f   // MARGIN + 0.05
#define LOTH 0.45f   // MARGIN - 0.05
#define BAND 0.10f   // UPTH - LOTH (in-band contribution)

typedef __attribute__((ext_vector_type(16))) _Float16 v16h;
typedef __attribute__((ext_vector_type(8)))  _Float16 v8h;
typedef __attribute__((ext_vector_type(4)))  _Float16 v4h;
typedef __attribute__((ext_vector_type(8)))  float    v8f;

// ---------------------------------------------------------------- init
__global__ void cl_init(float* __restrict__ rowT, float* __restrict__ rowC,
                        float* __restrict__ out, int n) {
    int i = blockIdx.x * blockDim.x + threadIdx.x;
    if (i < n) { rowT[i] = 0.0f; rowC[i] = 0.0f; }
    if (i == 0) out[0] = 0.0f;
}

// ------------------------------------------------- normalize + f16 pack
__global__ void cl_normalize(const float* __restrict__ x,
                             _Float16* __restrict__ xh, int n, int d) {
    int row  = blockIdx.x * (blockDim.x >> 5) + (threadIdx.x >> 5);
    int lane = threadIdx.x & 31;
    if (row >= n) return;

    const float4* src = (const float4*)(x + (size_t)row * d);
    int n4 = d >> 2;
    float4 vals[4];                        // supports D up to 512
    int cnt = 0;
    float ss = 0.0f;
    for (int c = lane; c < n4; c += 32) {
        float4 v = src[c];
        vals[cnt++] = v;
        ss += v.x * v.x + v.y * v.y + v.z * v.z + v.w * v.w;
    }
#pragma unroll
    for (int m = 16; m >= 1; m >>= 1) ss += __shfl_xor(ss, m, 32);
    float r = rsqrtf(ss);

    v4h* dst = (v4h*)(xh + (size_t)row * d);
    cnt = 0;
    for (int c = lane; c < n4; c += 32) {
        float4 v = vals[cnt++];
        v4h h;
        h[0] = (_Float16)(v.x * r);
        h[1] = (_Float16)(v.y * r);
        h[2] = (_Float16)(v.z * r);
        h[3] = (_Float16)(v.w * r);
        dst[c] = h;
    }
}

// Exact identity: for a negative pair,
//   max(up-v,0) + max(v-lo,0) == max3(up-v, v-lo, up-lo)   (v_max3_num_f32)
__device__ __forceinline__ float cl_term(float v, bool negpair) {
    float t = fmaxf(fmaxf(UPTH - v, v - LOTH), BAND);
    return negpair ? t : 0.0f;
}

// ---------------------------------------------- fused symmetric WMMA kernel
// DD = compile-time D (128); DD = 0 -> runtime d.
template <int DD>
__global__ void cl_wmma_tiles(const _Float16* __restrict__ xh,
                              const int* __restrict__ tgt,
                              float* __restrict__ rowT,
                              float* __restrict__ rowC,
                              int n, int d) {
    const int dlim = DD ? DD : d;
    int lane = threadIdx.x & 31;
    int wave = threadIdx.x >> 5;
    int im = blockIdx.y;                     // i macro-tile (32 rows)
    int jm = blockIdx.x * 8 + wave;          // j macro-tile (32 cols)
    int nm = n >> 5;
    if (jm >= nm || jm < im) return;         // upper triangle only; wave-uniform

    int lr = lane & 15;
    int hi = lane >> 4;
    bool mirror = (jm != im);

    const _Float16* arow0 = xh + (size_t)(im * 32 + lr) * dlim;   // A rows 0..15
    const _Float16* arow1 = arow0 + (size_t)16 * dlim;            // A rows 16..31
    const _Float16* brow0 = xh + (size_t)(jm * 32 + lr) * dlim;   // B rows 0..15
    const _Float16* brow1 = brow0 + (size_t)16 * dlim;            // B rows 16..31

    __builtin_prefetch(brow1 + dlim, 0, 1);  // global_prefetch_b8

    v8f acc00 = {}, acc01 = {}, acc10 = {}, acc11 = {};
#pragma unroll 1                             // rolled: keep fragments resident, no spills
    for (int k = 0; k < dlim; k += 32) {
        // A 16x32 f16 fragment: lanes 0-15 K={k..k+15},
        //                       lanes 16-31 K={k+16..k+31}
        v8h a0lo = *(const v8h*)(arow0 + k + hi * 16);
        v8h a0hi = *(const v8h*)(arow0 + k + 8 + hi * 16);
        v16h a0 = __builtin_shufflevector(a0lo, a0hi,
                  0, 1, 2, 3, 4, 5, 6, 7, 8, 9, 10, 11, 12, 13, 14, 15);
        v8h a1lo = *(const v8h*)(arow1 + k + hi * 16);
        v8h a1hi = *(const v8h*)(arow1 + k + 8 + hi * 16);
        v16h a1 = __builtin_shufflevector(a1lo, a1hi,
                  0, 1, 2, 3, 4, 5, 6, 7, 8, 9, 10, 11, 12, 13, 14, 15);
        // B 32x16 f16 fragment: lane holds 16 contiguous K of its column row
        v16h b0 = *(const v16h*)(brow0 + k + (hi << 4));
        v16h b1 = *(const v16h*)(brow1 + k + (hi << 4));

        acc00 = __builtin_amdgcn_wmma_f32_16x16x32_f16(false, a0, false, b0,
                                                       (short)0, acc00, false, false);
        acc01 = __builtin_amdgcn_wmma_f32_16x16x32_f16(false, a0, false, b1,
                                                       (short)0, acc01, false, false);
        acc10 = __builtin_amdgcn_wmma_f32_16x16x32_f16(false, a1, false, b0,
                                                       (short)0, acc10, false, false);
        acc11 = __builtin_amdgcn_wmma_f32_16x16x32_f16(false, a1, false, b1,
                                                       (short)0, acc11, false, false);
    }

    // ---------------- epilogue ----------------
    // D layout per tile: VGPR r holds (M = r + 8*hi, N = lane&15).
    // Ballot bits 0-15 = row (i0+r), bits 16-31 = row (i0+8+r), per tile.
    int i0 = im * 32, j0 = jm * 32;
    int jA = j0 + lr,      jB = j0 + 16 + lr;
    int tjA = tgt[jA],     tjB = tgt[jB];

    float colT_A = 0.0f, colT_B = 0.0f;      // column loss partials (this lane's col)
    int   colC_A = 0,    colC_B = 0;         // column count partials

#pragma unroll
    for (int r = 0; r < 8; ++r) {
        int iR0 = i0 + (hi << 3) + r;        // this lane's row in acc00/acc01
        int iR1 = iR0 + 16;                  // this lane's row in acc10/acc11
        int ti0 = tgt[iR0];
        int ti1 = tgt[iR1];
        bool n00 = (ti0 != tjA), n01 = (ti0 != tjB);
        bool n10 = (ti1 != tjA), n11 = (ti1 != tjB);
        float v00 = acc00[r], v01 = acc01[r], v10 = acc10[r], v11 = acc11[r];

        float t00 = cl_term(v00, n00);
        float t01 = cl_term(v01, n01);
        float t10 = cl_term(v10, n10);
        float t11 = cl_term(v11, n11);

        // validity counts as wave-wide bitmasks (strict v > lo, negatives only)
        unsigned b00 = (unsigned)__ballot(n00 && (v00 > LOTH));
        unsigned b01 = (unsigned)__ballot(n01 && (v01 > LOTH));
        unsigned b10 = (unsigned)__ballot(n10 && (v10 > LOTH));
        unsigned b11 = (unsigned)__ballot(n11 && (v11 > LOTH));

        // column partials (this lane's own element bit / term)
        colT_A += t00 + t10;
        colT_B += t01 + t11;
        colC_A += (int)((b00 >> lane) & 1u) + (int)((b10 >> lane) & 1u);
        colC_B += (int)((b01 >> lane) & 1u) + (int)((b11 >> lane) & 1u);

        // row loss partials: butterfly over the 16 columns (within each half-wave)
        float t0 = t00 + t01;                // row iR0
        float t1 = t10 + t11;                // row iR1
#pragma unroll
        for (int m = 1; m < 16; m <<= 1) {
            t0 += __shfl_xor(t0, m, 32);
            t1 += __shfl_xor(t1, m, 32);
        }
        // row counts: uniform popcounts of the ballot halves
        int c0lo = __popc(b00 & 0xffffu) + __popc(b01 & 0xffffu);  // row i0+r
        int c0hi = __popc(b00 >> 16)     + __popc(b01 >> 16);      // row i0+8+r
        int c1lo = __popc(b10 & 0xffffu) + __popc(b11 & 0xffffu);  // row i0+16+r
        int c1hi = __popc(b10 >> 16)     + __popc(b11 >> 16);      // row i0+24+r

        if (lr == 0) {                       // lane 0 -> low rows, lane 16 -> high rows
            atomicAdd(&rowT[iR0], t0);
            atomicAdd(&rowC[iR0], (float)(hi ? c0hi : c0lo));
            atomicAdd(&rowT[iR1], t1);
            atomicAdd(&rowC[iR1], (float)(hi ? c1hi : c1lo));
        }
    }

    if (mirror) {                            // credit symmetric lower-triangle rows
        colT_A += __shfl_xor(colT_A, 16, 32);
        colT_B += __shfl_xor(colT_B, 16, 32);
        colC_A += __shfl_xor(colC_A, 16, 32);
        colC_B += __shfl_xor(colC_B, 16, 32);
        if (hi == 0) {                       // lanes 0-15 commit column block A
            atomicAdd(&rowT[jA], colT_A);
            atomicAdd(&rowC[jA], (float)colC_A);
        } else {                             // lanes 16-31 commit column block B
            atomicAdd(&rowT[jB], colT_B);
            atomicAdd(&rowC[jB], (float)colC_B);
        }
    }
}

// ---------------------------------------------------------------- final
__global__ void cl_finalize(const float* __restrict__ rowT,
                            const float* __restrict__ rowC,
                            float* __restrict__ out, int n, float inv_n) {
    int i = blockIdx.x * blockDim.x + threadIdx.x;
    float v = 0.0f;
    if (i < n && rowC[i] >= 1.5f) v = rowT[i];   // need >= 2 valid negatives
#pragma unroll
    for (int m = 16; m >= 1; m >>= 1) v += __shfl_xor(v, m, 32);
    if ((threadIdx.x & 31) == 0) atomicAdd(out, v * inv_n);
}

// -------------------------------------------------------------- launch
extern "C" void kernel_launch(void* const* d_in, const int* in_sizes, int n_in,
                              void* d_out, int out_size, void* d_ws, size_t ws_size,
                              hipStream_t stream) {
    const float* x   = (const float*)d_in[0];
    const int*   tgt = (const int*)d_in[1];
    int n = in_sizes[1];                 // 8192
    int d = in_sizes[0] / n;             // 128
    float* out = (float*)d_out;

    // workspace: f16 normalized X (n*d*2 B) | rowT (n*4 B) | rowC (n*4 B)
    _Float16* xh   = (_Float16*)d_ws;
    float*    rowT = (float*)((char*)d_ws + (size_t)n * d * sizeof(_Float16));
    float*    rowC = rowT + n;

    cl_init<<<(n + 255) / 256, 256, 0, stream>>>(rowT, rowC, out, n);
    cl_normalize<<<(n + 7) / 8, 256, 0, stream>>>(x, xh, n, d);

    int nm = n >> 5;                     // 256 macro-tiles per dimension
    dim3 grid((nm + 7) / 8, nm);         // 8 j-macros (one per wave) per block
    if (d == 128)
        cl_wmma_tiles<128><<<grid, 256, 0, stream>>>(xh, tgt, rowT, rowC, n, d);
    else
        cl_wmma_tiles<0><<<grid, 256, 0, stream>>>(xh, tgt, rowT, rowC, n, d);

    cl_finalize<<<(n + 255) / 256, 256, 0, stream>>>(rowT, rowC, out, n, 1.0f / n);
}
